// AKOrNLayer_23270132810196
// MI455X (gfx1250) — compile-verified, matrix operand order checked by reading, not run on hardware
//
#include <hip/hip_runtime.h>
#include <hip/hip_bf16.h>
#include <math.h>

// Problem constants (match reference)
#define Bb 2
#define Nn 1024
#define Dd 256
#define Hh 4
#define Oo 8
#define DK 64
#define Gg (Bb*Hh)          // 8 (b,h) groups
#define Mm (Bb*Nn)          // 2048 rows
#define DTs 0.1f
#define NSTEPS 5
#define PI_F 3.14159265358979323846f
#define TWO_PI_F 6.28318530717958647692f

typedef __attribute__((ext_vector_type(16))) _Float16 v16h;
typedef __attribute__((ext_vector_type(8)))  _Float16 v8h;
typedef __attribute__((ext_vector_type(8)))  float    v8f;

// Build a 16x32 (16-bit, K=32) A/B WMMA fragment from a row-major f16 row.
// lanes 0-15: halves[0..7]=K(kb+0..7), halves[8..15]=K(kb+16..23)
// lanes 16-31: halves[0..7]=K(kb+8..15), halves[8..15]=K(kb+24..31)
__device__ __forceinline__ v16h load_frag(const _Float16* __restrict__ row, int kb, int lane) {
  const int off = (lane & 16) ? 8 : 0;
  v8h lo = *(const v8h*)(row + kb + off);
  v8h hi = *(const v8h*)(row + kb + 16 + off);
  v16h r;
#pragma unroll
  for (int t = 0; t < 8; ++t) { r[t] = lo[t]; r[8 + t] = hi[t]; }
  return r;
}

__global__ void cast_f16_kernel(const float* __restrict__ in, _Float16* __restrict__ out, int n) {
  int i = blockIdx.x * blockDim.x + threadIdx.x;
  if (i < n) out[i] = (_Float16)in[i];
}

// q/k projection: C(16x16) = x(16x256) @ W^T tile; output f16 laid out [g][n][dk]
__global__ void proj_qk_kernel(const _Float16* __restrict__ X,  // [2048][256]
                               const _Float16* __restrict__ W,  // [256][256] row-major (out-feature rows)
                               _Float16* __restrict__ out) {    // [G][N][DK]
  const int lane = threadIdx.x;
  const int m0 = blockIdx.x * 16, d0 = blockIdx.y * 16;
  const _Float16* xrow = X + (size_t)(m0 + (lane & 15)) * Dd;
  const _Float16* wrow = W + (size_t)(d0 + (lane & 15)) * Dd;
  v8f acc = {};
#pragma unroll
  for (int kb = 0; kb < Dd; kb += 32) {
    v16h a = load_frag(xrow, kb, lane);
    v16h b = load_frag(wrow, kb, lane);
    acc = __builtin_amdgcn_wmma_f32_16x16x32_f16(false, a, false, b, (short)0, acc, false, false);
  }
  // C frag: lane 0-15 -> col d0+lane, rows m0+0..7 ; lane 16-31 -> col d0+lane-16, rows m0+8..15
  const int d = d0 + (lane & 15);
  const int h = d / DK, d64 = d % DK;
  const int g = (m0 / Nn) * Hh + h;
  const int nbase = (m0 % Nn) + ((lane & 16) ? 8 : 0);
#pragma unroll
  for (int r = 0; r < 8; ++r)
    out[((size_t)g * Nn + (nbase + r)) * DK + d64] = (_Float16)acc[r];
}

// phase projection: phases = x @ Wp^T + bp, fp32, laid out [G][N][O]
__global__ void proj_p_kernel(const _Float16* __restrict__ X,   // [2048][256]
                              const _Float16* __restrict__ Wp,  // [32][256]
                              const float* __restrict__ bp,     // [32]
                              float* __restrict__ ph) {         // [G][N][O]
  const int lane = threadIdx.x;
  const int m0 = blockIdx.x * 16, d0 = blockIdx.y * 16;
  const _Float16* xrow = X + (size_t)(m0 + (lane & 15)) * Dd;
  const _Float16* wrow = Wp + (size_t)(d0 + (lane & 15)) * Dd;
  v8f acc = {};
#pragma unroll
  for (int kb = 0; kb < Dd; kb += 32) {
    v16h a = load_frag(xrow, kb, lane);
    v16h b = load_frag(wrow, kb, lane);
    acc = __builtin_amdgcn_wmma_f32_16x16x32_f16(false, a, false, b, (short)0, acc, false, false);
  }
  const int d = d0 + (lane & 15);        // 0..31
  const int h = d / Oo, o = d % Oo;
  const int g = (m0 / Nn) * Hh + h;
  const int nbase = (m0 % Nn) + ((lane & 16) ? 8 : 0);
  const float bias = bp[d];
#pragma unroll
  for (int r = 0; r < 8; ++r)
    ph[((size_t)g * Nn + (nbase + r)) * Oo + o] = acc[r] + bias;
}

// V^T table: Vt[g][vcol][j], vcol 0..7 = sin(ph[.,o]), 8..15 = cos(ph[.,o])
__global__ void sincos_v_kernel(const float* __restrict__ ph, _Float16* __restrict__ Vt) {
  int idx = blockIdx.x * blockDim.x + threadIdx.x;   // over G*N
  if (idx >= Gg * Nn) return;
  const int g = idx >> 10, j = idx & (Nn - 1);
  const float* p = ph + (size_t)idx * Oo;
  _Float16* v = Vt + (size_t)g * 16 * Nn + j;
#pragma unroll
  for (int o = 0; o < Oo; ++o) {
    float ang = p[o];
    v[(size_t)o * Nn]        = (_Float16)sinf(ang);
    v[(size_t)(8 + o) * Nn]  = (_Float16)cosf(ang);
  }
}

// One Kuramoto step: flash softmax-attention J@[s|c] + in-place phase update.
// grid: (G, N/16), 32 threads (one wave).
__global__ void kuramoto_step_kernel(const _Float16* __restrict__ qh,   // [G][N][DK]
                                     const _Float16* __restrict__ kh,   // [G][N][DK]
                                     const _Float16* __restrict__ Vt,   // [G][16][N]
                                     float* __restrict__ ph,            // [G][N][O]
                                     const float* __restrict__ omega,   // [H][O]
                                     const float* __restrict__ cscale) {
  const int lane = threadIdx.x;
  const int g = blockIdx.x;
  const int h = g % Hh;
  const int i0 = blockIdx.y * 16;
  const _Float16* qg = qh + (size_t)g * Nn * DK;
  const _Float16* kg = kh + (size_t)g * Nn * DK;
  const _Float16* vg = Vt + (size_t)g * 16 * Nn;

  // q^T as B fragments for dk chunks {0,32}: lane n -> row i0+n
  const _Float16* qrow = qg + (size_t)(i0 + (lane & 15)) * DK;
  const v16h qf0 = load_frag(qrow, 0, lane);
  const v16h qf1 = load_frag(qrow, 32, lane);
  // V^T A fragment source row: M = vcol = lane&15
  const _Float16* vrow = vg + (size_t)(lane & 15) * Nn;

  v8f acc = {};                 // U^T accumulator: lane n holds i=n; regs vcols
  float m_run = -3.0e38f, z_run = 0.0f;
  const float scale = 0.125f;   // 1/sqrt(64)

  for (int j0 = 0; j0 < Nn; j0 += 32) {
    // S^T = K_tile @ q^T for 32 keys as two 16-key tiles
    const _Float16* krow0 = kg + (size_t)(j0 + (lane & 15)) * DK;
    const _Float16* krow1 = krow0 + 16 * DK;
    v16h ka0  = load_frag(krow0, 0,  lane);
    v16h ka0b = load_frag(krow0, 32, lane);
    v16h ka1  = load_frag(krow1, 0,  lane);
    v16h ka1b = load_frag(krow1, 32, lane);
    v8f s0 = {}, s1 = {};
    s0 = __builtin_amdgcn_wmma_f32_16x16x32_f16(false, ka0,  false, qf0, (short)0, s0, false, false);
    s0 = __builtin_amdgcn_wmma_f32_16x16x32_f16(false, ka0b, false, qf1, (short)0, s0, false, false);
    s1 = __builtin_amdgcn_wmma_f32_16x16x32_f16(false, ka1,  false, qf0, (short)0, s1, false, false);
    s1 = __builtin_amdgcn_wmma_f32_16x16x32_f16(false, ka1b, false, qf1, (short)0, s1, false, false);

    float mx = -3.0e38f;
#pragma unroll
    for (int r = 0; r < 8; ++r) {
      s0[r] *= scale; s1[r] *= scale;
      mx = fmaxf(mx, fmaxf(s0[r], s1[r]));
    }
    mx = fmaxf(mx, __shfl_xor(mx, 16, 32));            // lanes n / n+16 share row i=n
    const float m_new = fmaxf(m_run, mx);
    const float resc = __expf(m_run - m_new);
    float zsum = 0.0f;
    v16h pf;                                           // P^T as B fragment (register-only remap)
#pragma unroll
    for (int r = 0; r < 8; ++r) {
      float e0 = __expf(s0[r] - m_new);
      float e1 = __expf(s1[r] - m_new);
      zsum += e0 + e1;
      pf[r] = (_Float16)e0;
      pf[8 + r] = (_Float16)e1;
    }
    zsum += __shfl_xor(zsum, 16, 32);
    z_run = z_run * resc + zsum;
    m_run = m_new;

    v16h vf = load_frag(vrow, j0, lane);               // V^T A fragment, keys j0..j0+31
#pragma unroll
    for (int r = 0; r < 8; ++r) acc[r] *= resc;
    acc = __builtin_amdgcn_wmma_f32_16x16x32_f16(false, vf, false, pf, (short)0, acc, false, false);
  }

  const float invz = 1.0f / z_run;
  float js[8], jc[8];
#pragma unroll
  for (int r = 0; r < 8; ++r) {
    float u = acc[r] * invz;     // lane<16: (J s)[i][r], lane>=16: (J c)[i][r]
    js[r] = u;
    jc[r] = __shfl_down(u, 16, 32);
  }
  if ((lane & 16) == 0) {
    const int i = i0 + lane;
    float* prow = ph + ((size_t)g * Nn + i) * Oo;
    const float cs = cscale[0];
#pragma unroll
    for (int o = 0; o < Oo; ++o) {
      float p = prow[o];
      float sv = sinf(p), cv = cosf(p);
      float coup = cv * js[o] - sv * jc[o];
      float nw = p + DTs * (omega[h * Oo + o] + cs * coup);
      nw = fmodf(nw + PI_F, TWO_PI_F);
      if (nw < 0.0f) nw += TWO_PI_F;
      prow[o] = nw - PI_F;
    }
  }
}

// sig[b][n][h*8+o] = cos(ph[(b,h)][n][o]) as f16
__global__ void make_sig_kernel(const float* __restrict__ ph, _Float16* __restrict__ sig) {
  int idx = blockIdx.x * blockDim.x + threadIdx.x;   // over B*N
  if (idx >= Mm) return;
  const int b = idx >> 10, n = idx & (Nn - 1);
  _Float16* srow = sig + (size_t)idx * (Hh * Oo);
#pragma unroll
  for (int h = 0; h < Hh; ++h) {
    const float* p = ph + ((size_t)(b * Hh + h) * Nn + n) * Oo;
#pragma unroll
    for (int o = 0; o < Oo; ++o) srow[h * Oo + o] = (_Float16)cosf(p[o]);
  }
}

// out = sig @ Wo^T + bo  (M=2048, N=256, K=32 -> one WMMA per tile)
__global__ void out_gemm_kernel(const _Float16* __restrict__ sig,  // [2048][32]
                                const _Float16* __restrict__ Wo,   // [256][32]
                                const float* __restrict__ bo,      // [256]
                                float* __restrict__ out) {         // [2048][256]
  const int lane = threadIdx.x;
  const int m0 = blockIdx.x * 16, d0 = blockIdx.y * 16;
  const _Float16* arow = sig + (size_t)(m0 + (lane & 15)) * 32;
  const _Float16* brow = Wo  + (size_t)(d0 + (lane & 15)) * 32;
  v16h a = load_frag(arow, 0, lane);
  v16h b = load_frag(brow, 0, lane);
  v8f acc = {};
  acc = __builtin_amdgcn_wmma_f32_16x16x32_f16(false, a, false, b, (short)0, acc, false, false);
  const int d = d0 + (lane & 15);
  const float bias = bo[d];
  const int mbase = m0 + ((lane & 16) ? 8 : 0);
#pragma unroll
  for (int r = 0; r < 8; ++r)
    out[(size_t)(mbase + r) * Dd + d] = acc[r] + bias;
}

extern "C" void kernel_launch(void* const* d_in, const int* in_sizes, int n_in,
                              void* d_out, int out_size, void* d_ws, size_t ws_size,
                              hipStream_t stream) {
  const float* x   = (const float*)d_in[0];   // [2][1024][256]
  const float* Wq  = (const float*)d_in[1];   // [256][256]
  const float* Wk  = (const float*)d_in[2];   // [256][256]
  const float* Wp  = (const float*)d_in[3];   // [32][256]
  const float* bp  = (const float*)d_in[4];   // [32]
  const float* Wo  = (const float*)d_in[5];   // [256][32]
  const float* bo  = (const float*)d_in[6];   // [256]
  const float* om  = (const float*)d_in[7];   // [4][8]
  const float* cs  = (const float*)d_in[8];   // scalar
  float* out = (float*)d_out;

  // workspace carving (all 256B aligned)
  char* w = (char*)d_ws;
  size_t off = 0;
  auto take = [&](size_t bytes) { char* p = w + off; off = (off + bytes + 255) & ~(size_t)255; return p; };
  _Float16* xh  = (_Float16*)take((size_t)Mm * Dd * 2);
  _Float16* Wqh = (_Float16*)take((size_t)Dd * Dd * 2);
  _Float16* Wkh = (_Float16*)take((size_t)Dd * Dd * 2);
  _Float16* Wph = (_Float16*)take((size_t)32 * Dd * 2);
  _Float16* Woh = (_Float16*)take((size_t)Dd * 32 * 2);
  _Float16* qh  = (_Float16*)take((size_t)Gg * Nn * DK * 2);
  _Float16* kh  = (_Float16*)take((size_t)Gg * Nn * DK * 2);
  float*    ph  = (float*)   take((size_t)Gg * Nn * Oo * 4);
  _Float16* Vt  = (_Float16*)take((size_t)Gg * 16 * Nn * 2);
  _Float16* sig = (_Float16*)take((size_t)Mm * 32 * 2);

  // 1. f16 conversions
  cast_f16_kernel<<<(Mm * Dd + 255) / 256, 256, 0, stream>>>(x, xh, Mm * Dd);
  cast_f16_kernel<<<(Dd * Dd + 255) / 256, 256, 0, stream>>>(Wq, Wqh, Dd * Dd);
  cast_f16_kernel<<<(Dd * Dd + 255) / 256, 256, 0, stream>>>(Wk, Wkh, Dd * Dd);
  cast_f16_kernel<<<(32 * Dd + 255) / 256, 256, 0, stream>>>(Wp, Wph, 32 * Dd);
  cast_f16_kernel<<<(Dd * 32 + 255) / 256, 256, 0, stream>>>(Wo, Woh, Dd * 32);

  // 2. projections (WMMA)
  proj_qk_kernel<<<dim3(Mm / 16, Dd / 16), 32, 0, stream>>>(xh, Wqh, qh);
  proj_qk_kernel<<<dim3(Mm / 16, Dd / 16), 32, 0, stream>>>(xh, Wkh, kh);
  proj_p_kernel<<<dim3(Mm / 16, 2), 32, 0, stream>>>(xh, Wph, bp, ph);

  // 3. 5 Kuramoto steps (V snapshot, then flash-attention coupling + update)
  for (int s = 0; s < NSTEPS; ++s) {
    sincos_v_kernel<<<(Gg * Nn + 255) / 256, 256, 0, stream>>>(ph, Vt);
    kuramoto_step_kernel<<<dim3(Gg, Nn / 16), 32, 0, stream>>>(qh, kh, Vt, ph, om, cs);
  }

  // 4. readout
  make_sig_kernel<<<(Mm + 255) / 256, 256, 0, stream>>>(ph, sig);
  out_gemm_kernel<<<dim3(Mm / 16, Dd / 16), 32, 0, stream>>>(sig, Woh, bo, out);
}
